// ForwardDeformer_50714973831481
// MI455X (gfx1250) — matrix operand order repaired and checked by reading, not compiled, not run on hardware
//
#include <hip/hip_runtime.h>
#include <hip/hip_bf16.h>

// ForwardDeformer (SNARF-style Broyden root-find) for MI455X / gfx1250.
// Output == x_opt from _solve (the reference "correction" term is identically
// zero in value). All MLP GEMMs run on V_WMMA_F32_16X16X32_F16.
//
// Round-5 changes: vectorize the remaining scalar LDS traffic (tfs rows,
// W0/B0 in layer 0, f16 delta row in the dx reduction) to b128 loads; all
// touched regions are 16B-aligned by construction.

typedef __attribute__((ext_vector_type(16))) _Float16     v16h;
typedef __attribute__((ext_vector_type(8)))  float        v8f;
typedef __attribute__((ext_vector_type(4)))  float        v4f;
typedef __attribute__((ext_vector_type(8)))  unsigned int v8u;
typedef __attribute__((ext_vector_type(4)))  unsigned int v4u;

union HU { v16h h; v8u u; };
union HV { v4u u; _Float16 hh[8]; };

#define WAVES   4
#define NPTS    (2048 * 45)
#define BLOCKS  (NPTS / (16 * WAVES))   // 1440

#define WSTR    136   // padded row stride (f16 elems) for 128-wide tiles (272B)
#define DSTR    72    // padded row stride for the 64-wide cotangent tile (144B)

#define LOG2E   1.4426950408889634f
#define LN2     0.6931471805599453f

// ---------- small scalar helpers ----------
static __device__ __forceinline__ float fexp(float v) {      // e^v, v <= 0 here
  return __builtin_amdgcn_exp2f(v * LOG2E);
}

// softplus(100x)/100, branchless & fixup-free:
// softplus(t) = max(t,0) + log(1 + exp(-|t|)); log2 arg is in (1,2].
static __device__ __forceinline__ float actfn(float x) {
  float t = 100.f * x;
  float z = __builtin_amdgcn_exp2f(-fabsf(t) * LOG2E);
  float l = __builtin_amdgcn_logf(1.f + z) * LN2;
  return (fmaxf(t, 0.f) + l) * 0.01f;
}

static __device__ __forceinline__ void inv3(const float A[9], float O[9]) {
  float a=A[0],b=A[1],c=A[2],d=A[3],e=A[4],f=A[5],g=A[6],h=A[7],i=A[8];
  float C00=e*i-f*h, C01=c*h-b*i, C02=b*f-c*e;
  float C10=f*g-d*i, C11=a*i-c*g, C12=c*d-a*f;
  float C20=d*h-e*g, C21=b*g-a*h, C22=a*e-b*d;
  float det = a*C00 + b*C10 + c*C20;
  float rd = 1.f / det;
  O[0]=C00*rd; O[1]=C01*rd; O[2]=C02*rd;
  O[3]=C10*rd; O[4]=C11*rd; O[5]=C12*rd;
  O[6]=C20*rd; O[7]=C21*rd; O[8]=C22*rd;
}

// ---------- WMMA fragment helpers ----------
static __device__ __forceinline__ v8f wmma16(v16h a, v16h b, v8f c) {
  return __builtin_amdgcn_wmma_f32_16x16x32_f16(false, a, false, b, (short)0, c,
                                                false, false);
}

// A-fragment: 16x32 f16 tile, row-major (row stride = STRIDE f16 elems).
// Two 16B-aligned 16B chunks -> two ds_load_b128.
template<int STRIDE>
static __device__ __forceinline__ v16h load_A(const _Float16* buf, int kb, int lane) {
  int m  = lane & 15;
  int hi = lane >> 4;
  const _Float16* base = buf + m * STRIDE + kb * 32 + hi * 16;
  v4u q0 = *(const v4u*)(base);
  v4u q1 = *(const v4u*)(base + 8);
  HU r;
#pragma unroll
  for (int i = 0; i < 4; ++i) { r.u[i] = q0[i]; r.u[i+4] = q1[i]; }
  return r.h;
}

// Forward B-fragment from TRANSPOSED weights Wt[n][k] (row stride WSTR):
// element (k,n) = Wt[n*WSTR+k]; 32 contiguous bytes -> two ds_load_b128.
static __device__ __forceinline__ v16h load_Bf(const _Float16* Wt, int kb, int nb, int lane) {
  int n  = (lane & 15) + nb * 16;
  int hi = lane >> 4;
  const _Float16* row = Wt + n * WSTR + kb * 32 + hi * 16;
  v4u q0 = *(const v4u*)(row);
  v4u q1 = *(const v4u*)(row + 8);
  HU r;
#pragma unroll
  for (int i = 0; i < 4; ++i) { r.u[i] = q0[i]; r.u[i+4] = q1[i]; }
  return r.h;
}

// Backward B-fragment (B = W^T): element (k,n) = W[n][k] = Wt[k*WSTR+n] (strided).
static __device__ __forceinline__ v16h load_Bb(const _Float16* Wt, int kb, int nb, int lane) {
  int n  = (lane & 15) + nb * 16;
  int hi = lane >> 4;
  v16h b;
#pragma unroll
  for (int v = 0; v < 8; ++v) {
    int k0 = kb * 32 + hi * 16 + 2 * v;
    b[2*v]   = Wt[k0 * WSTR + n];
    b[2*v+1] = Wt[(k0 + 1) * WSTR + n];
  }
  return b;
}

// ---------- per-wave shared-memory view ----------
struct Shm {
  _Float16 *W1t, *W2t, *W3t, *WoT;  // transposed f16 weights (LDS, whole block)
  _Float16 *act;                    // [4 layers][16][WSTR] f16, this wave
  _Float16 *delta;                  // [16][WSTR] f16, this wave
  float *W0, *B0, *B1, *B2, *B3, *Bo;
  float *tfs;                       // [55][12] rows 0..2 of each 4x4 (16B/row-aligned)
  float *tinv;                      // [45][12] inverse affines
  float *logit;                     // [16][64] f32 (logits -> softmax weights)
  float *xsh;                       // [16][4]  x_h per point
};

// out = act(in @ W + bias), in/out are [16][WSTR] f16, Wt transposed
static __device__ __forceinline__ void fwd_layer(const _Float16* in, const _Float16* Wt,
                                                 const float* bias, _Float16* out, int lane) {
  v16h a[4];
#pragma unroll
  for (int kb = 0; kb < 4; ++kb) a[kb] = load_A<WSTR>(in, kb, lane);
#pragma unroll
  for (int nb = 0; nb < 8; ++nb) {
    v8f c = {0.f,0.f,0.f,0.f,0.f,0.f,0.f,0.f};
#pragma unroll
    for (int kb = 0; kb < 4; ++kb) {
      v16h b = load_Bf(Wt, kb, nb, lane);
      c = wmma16(a[kb], b, c);
    }
    int n = nb * 16 + (lane & 15);
    float bv = bias[n];
#pragma unroll
    for (int v = 0; v < 8; ++v) {
      int m = (lane >> 4) * 8 + v;
      out[m * WSTR + n] = (_Float16)actfn(c[v] + bv);
    }
  }
}

// logits[16][64] = in[16][128] @ Wo[128][64] + bo  (WoT = [64][WSTR])
static __device__ __forceinline__ void out_layer(const _Float16* in, const _Float16* WoT,
                                                 const float* bo, float* logit, int lane) {
  v16h a[4];
#pragma unroll
  for (int kb = 0; kb < 4; ++kb) a[kb] = load_A<WSTR>(in, kb, lane);
#pragma unroll
  for (int nb = 0; nb < 4; ++nb) {
    v8f c = {0.f,0.f,0.f,0.f,0.f,0.f,0.f,0.f};
#pragma unroll
    for (int kb = 0; kb < 4; ++kb) {
      v16h b = load_Bf(WoT, kb, nb, lane);
      c = wmma16(a[kb], b, c);
    }
    int n = nb * 16 + (lane & 15);
    float bv = bo[n];
#pragma unroll
    for (int v = 0; v < 8; ++v) {
      int m = (lane >> 4) * 8 + v;
      logit[m * 64 + n] = c[v] + bv;
    }
  }
}

// dout[16][WSTR] = (din[16][K] @ W^T) * act'(pre); act' = 1 - exp(-100*a).
// In-place safe: all A fragments hoisted before any store (wave lockstep).
template<int KSTEPS, int INSTRIDE>
static __device__ __forceinline__ void bwd_layer(const _Float16* din, const _Float16* Wt,
                                                 const _Float16* actL, _Float16* dout, int lane) {
  v16h a[KSTEPS];
#pragma unroll
  for (int kb = 0; kb < KSTEPS; ++kb) a[kb] = load_A<INSTRIDE>(din, kb, lane);
#pragma unroll
  for (int nb = 0; nb < 8; ++nb) {
    v8f c = {0.f,0.f,0.f,0.f,0.f,0.f,0.f,0.f};
#pragma unroll
    for (int kb = 0; kb < KSTEPS; ++kb) {
      v16h b = load_Bb(Wt, kb, nb, lane);
      c = wmma16(a[kb], b, c);
    }
    int n = nb * 16 + (lane & 15);
#pragma unroll
    for (int v = 0; v < 8; ++v) {
      int m = (lane >> 4) * 8 + v;
      float av = (float)actL[m * WSTR + n];
      float sg = 1.f - fexp(-100.f * av);
      dout[m * WSTR + n] = (_Float16)(c[v] * sg);
    }
  }
}

// g(x) = forward_skinning(x) - xd_target for this wave's 16 points.
// Bones split across lane pairs: lane m handles bones 0-27 of point m,
// lane m+16 handles bones 28-54; partials combined with __shfl_xor(.,16).
static __device__ __forceinline__ void g_eval(const Shm& S, int lane,
                                              float x0, float x1, float x2,
                                              float tx, float ty, float tz,
                                              float& e0, float& e1, float& e2,
                                              float& o0, float& o1, float& o2) {
  if (lane < 16) {
    S.xsh[lane*4+0] = x0; S.xsh[lane*4+1] = x1;
    S.xsh[lane*4+2] = x2; S.xsh[lane*4+3] = 1.f;
  }
  __syncthreads();
  {
    // layer 0 (K=3) on VALU: lane pair (2m,2m+1) produces row m's 128 outputs
    int m = lane >> 1;
    int hbase = (lane & 1) * 64;
    float xa = S.xsh[m*4+0], xb = S.xsh[m*4+1], xc = S.xsh[m*4+2];
    for (int t = 0; t < 64; t += 4) {
      int h = hbase + t;
      v4f w0 = *(const v4f*)(S.W0 + h);
      v4f w1 = *(const v4f*)(S.W0 + 128 + h);
      v4f w2 = *(const v4f*)(S.W0 + 256 + h);
      v4f bb = *(const v4f*)(S.B0 + h);
#pragma unroll
      for (int q = 0; q < 4; ++q) {
        float pre = xa*w0[q] + xb*w1[q] + xc*w2[q] + bb[q];
        S.act[m*WSTR + h + q] = (_Float16)actfn(pre);
      }
    }
  }
  __syncthreads();
  fwd_layer(S.act,           S.W1t, S.B1, S.act + 16*WSTR,   lane);
  fwd_layer(S.act + 16*WSTR, S.W2t, S.B2, S.act + 32*WSTR,   lane);
  fwd_layer(S.act + 32*WSTR, S.W3t, S.B3, S.act + 48*WSTR,   lane);
  out_layer(S.act + 48*WSTR, S.WoT, S.Bo, S.logit, lane);
  __syncthreads();

  const int m   = lane & 15;
  const int nb0 = (lane >> 4) * 28;      // bone range base: 0 or 28
  // softmax over 55 bones, split across the lane pair
  float mxh = -3.4e38f;
  for (int j = 0; j < 28; ++j) {
    int nn = nb0 + j;
    if (nn < 55) mxh = fmaxf(mxh, S.logit[m*64 + nn]);
  }
  float mx = fmaxf(mxh, __shfl_xor(mxh, 16, 32));
  float sh = 0.f;
  for (int j = 0; j < 28; ++j) {
    int nn = nb0 + j;
    if (nn < 55) {
      float e = fexp(20.f * (S.logit[m*64 + nn] - mx));
      S.logit[m*64 + nn] = e; sh += e;
    }
  }
  float inv = 1.f / (sh + __shfl_xor(sh, 16, 32));
  float p0 = 0.f, p1 = 0.f, p2 = 0.f;
  for (int j = 0; j < 28; ++j) {
    int nn = nb0 + j;
    if (nn < 55) {
      float w = S.logit[m*64 + nn] * inv;
      S.logit[m*64 + nn] = w;                // keep softmax weights
      const v4f* Av = (const v4f*)(S.tfs + nn * 12);   // 3x b128
      v4f r0 = Av[0], r1 = Av[1], r2 = Av[2];
      p0 += w * (r0[0]*x0 + r0[1]*x1 + r0[2]*x2 + r0[3]);
      p1 += w * (r1[0]*x0 + r1[1]*x1 + r1[2]*x2 + r1[3]);
      p2 += w * (r2[0]*x0 + r2[1]*x1 + r2[2]*x2 + r2[3]);
    }
  }
  o0 = p0 + __shfl_xor(p0, 16, 32);
  o1 = p1 + __shfl_xor(p1, 16, 32);
  o2 = p2 + __shfl_xor(p2, 16, 32);
  e0 = o0 - tx; e1 = o1 - ty; e2 = o2 - tz;
  __syncthreads();
}

__global__ __launch_bounds__(128)
void deformer_kernel(const float* __restrict__ xd,  const float* __restrict__ tfs_g,
                     const float* __restrict__ W0g, const float* __restrict__ b0g,
                     const float* __restrict__ W1g, const float* __restrict__ b1g,
                     const float* __restrict__ W2g, const float* __restrict__ b2g,
                     const float* __restrict__ W3g, const float* __restrict__ b3g,
                     const float* __restrict__ Wog, const float* __restrict__ bog,
                     float* __restrict__ out) {
  extern __shared__ char smem[];
  _Float16* sW1t      = (_Float16*)smem;           // [128][WSTR] transposed
  _Float16* sW2t      = sW1t + 128*WSTR;
  _Float16* sW3t      = sW2t + 128*WSTR;
  _Float16* sWoT      = sW3t + 128*WSTR;           // [64][WSTR] transposed, padded
  _Float16* sActAll   = sWoT + 64*WSTR;            // WAVES*4*16*WSTR
  _Float16* sDeltaAll = sActAll + WAVES*4*16*WSTR; // WAVES*16*WSTR
  float* fbase  = (float*)(sDeltaAll + WAVES*16*WSTR);
  float* sW0    = fbase;            // 3*128
  float* sB0    = sW0 + 384;
  float* sB1    = sB0 + 128;
  float* sB2    = sB1 + 128;
  float* sB3    = sB2 + 128;
  float* sBo    = sB3 + 128;        // 64
  float* sTfs   = sBo + 64;         // 55*12  (rows 48B -> 16B aligned)
  float* sTinv  = sTfs + 660;       // 45*12
  float* sLogit = sTinv + 540;      // WAVES*16*64
  float* sX     = sLogit + WAVES*1024; // WAVES*16*4

  const int tid = threadIdx.x, lane = tid & 31, wave = tid >> 5;

  // ---- stage weights (transposed, padded) / transforms into LDS ----
  for (int i = tid; i < 128*128; i += 128) {
    int r = i >> 7, c = i & 127;                 // Wt[r][c] = W[c][r]
    sW1t[r*WSTR + c] = (_Float16)W1g[c*128 + r];
    sW2t[r*WSTR + c] = (_Float16)W2g[c*128 + r];
    sW3t[r*WSTR + c] = (_Float16)W3g[c*128 + r];
  }
  for (int i = tid; i < 64*128; i += 128) {
    int r = i >> 7, c = i & 127;                 // WoT[r][c] = Wo[c][r]
    sWoT[r*WSTR + c] = (_Float16)((r < 55) ? Wog[c*55 + r] : 0.f);
  }
  for (int i = tid; i < 384; i += 128) sW0[i] = W0g[i];
  if (tid < 128) { sB0[tid]=b0g[tid]; sB1[tid]=b1g[tid]; sB2[tid]=b2g[tid]; sB3[tid]=b3g[tid]; }
  if (tid < 64)  sBo[tid] = (tid < 55) ? bog[tid] : 0.f;
  if (tid < 55) {
#pragma unroll
    for (int d = 0; d < 12; ++d) sTfs[tid*12 + d] = tfs_g[tid*16 + d];
  }
  if (tid < 45) {  // INIT_BONES mapping + affine inverse
    int bone = (tid < 3) ? tid : ((tid < 5) ? tid + 1 : tid + 10);
    const float* T = tfs_g + bone * 16;
    float R[9] = {T[0],T[1],T[2], T[4],T[5],T[6], T[8],T[9],T[10]};
    float Ri[9]; inv3(R, Ri);
    float t0 = T[3], t1 = T[7], t2 = T[11];
    float* o = sTinv + tid * 12;
    o[0]=Ri[0]; o[1]=Ri[1]; o[2]=Ri[2];  o[3]  = -(Ri[0]*t0 + Ri[1]*t1 + Ri[2]*t2);
    o[4]=Ri[3]; o[5]=Ri[4]; o[6]=Ri[5];  o[7]  = -(Ri[3]*t0 + Ri[4]*t1 + Ri[5]*t2);
    o[8]=Ri[6]; o[9]=Ri[7]; o[10]=Ri[8]; o[11] = -(Ri[6]*t0 + Ri[7]*t1 + Ri[8]*t2);
  }
  __syncthreads();

  Shm S;
  S.W1t = sW1t; S.W2t = sW2t; S.W3t = sW3t; S.WoT = sWoT;
  S.act   = sActAll   + wave * 4 * 16 * WSTR;
  S.delta = sDeltaAll + wave * 16 * WSTR;
  S.W0 = sW0; S.B0 = sB0; S.B1 = sB1; S.B2 = sB2; S.B3 = sB3; S.Bo = sBo;
  S.tfs = sTfs; S.tinv = sTinv;
  S.logit = sLogit + wave * 1024;
  S.xsh   = sX + wave * 64;

  // lanes 16-31 mirror lanes 0-15 per-point state; bone work is split
  const int p  = (blockIdx.x * WAVES + wave) * 16 + (lane & 15);
  const int nn_pt = p / 45, ii = p % 45;
  const float tx = xd[nn_pt*3+0], ty = xd[nn_pt*3+1], tz = xd[nn_pt*3+2];
  const float* Ti = sTinv + ii * 12;
  float x0 = Ti[0]*tx + Ti[1]*ty + Ti[2]*tz  + Ti[3];
  float x1 = Ti[4]*tx + Ti[5]*ty + Ti[6]*tz  + Ti[7];
  float x2 = Ti[8]*tx + Ti[9]*ty + Ti[10]*tz + Ti[11];

  // ---- forward at xc_init (also fills act buffers + softmax weights) ----
  float e0, e1, e2, o0, o1, o2;
  g_eval(S, lane, x0, x1, x2, tx, ty, tz, e0, e1, e2, o0, o1, o2);

  // ---- exact 3x3 Jacobian via 3 VJP passes, then invert -> J_inv0 ----
  const int m   = lane & 15;
  const int hi  = lane >> 4;
  const int nb0 = hi * 28;
  float Jm[9] = {1.f,0.f,0.f, 0.f,1.f,0.f, 0.f,0.f,1.f};
#pragma unroll
  for (int i = 0; i < 3; ++i) {
    {
      float oi = (i == 0) ? o0 : ((i == 1) ? o1 : o2);
      for (int j = 0; j < 28; ++j) {        // cotangent in logit space, K=64
        int nn = nb0 + j;
        if (nn < 55) {
          v4f ri = ((const v4f*)(S.tfs + nn * 12))[i];   // row i: 1x b128
          float w = S.logit[m*64 + nn];
          float y = ri[0]*x0 + ri[1]*x1 + ri[2]*x2 + ri[3];
          S.delta[m*DSTR + nn] = (_Float16)(20.f * w * (y - oi));
        }
      }
      if (hi) {                             // zero padding lanes 55..63
        for (int nn = 55; nn < 64; ++nn) S.delta[m*DSTR + nn] = (_Float16)0.f;
      }
    }
    __syncthreads();
    bwd_layer<2, DSTR>(S.delta, S.WoT, S.act + 48*WSTR, S.delta, lane);
    bwd_layer<4, WSTR>(S.delta, S.W3t, S.act + 32*WSTR, S.delta, lane);
    bwd_layer<4, WSTR>(S.delta, S.W2t, S.act + 16*WSTR, S.delta, lane);
    bwd_layer<4, WSTR>(S.delta, S.W1t, S.act,           S.delta, lane);
    __syncthreads();
    {
      // dx = dh0 @ W0^T, h range split 64/64 across the lane pair; b128 reads
      float dxa = 0.f, dxb = 0.f, dxc = 0.f;
      const _Float16* drow = S.delta + m*WSTR + hi*64;
      for (int jb = 0; jb < 64; jb += 8) {
        HV hv; hv.u = *(const v4u*)(drow + jb);
        int h = hi*64 + jb;
        v4f a0 = *(const v4f*)(S.W0 + h),        a1 = *(const v4f*)(S.W0 + h + 4);
        v4f b0 = *(const v4f*)(S.W0 + 128 + h),  b1 = *(const v4f*)(S.W0 + 132 + h);
        v4f c0 = *(const v4f*)(S.W0 + 256 + h),  c1 = *(const v4f*)(S.W0 + 260 + h);
#pragma unroll
        for (int q = 0; q < 4; ++q) {
          float dlo = (float)hv.hh[q], dhi = (float)hv.hh[q+4];
          dxa += dlo*a0[q] + dhi*a1[q];
          dxb += dlo*b0[q] + dhi*b1[q];
          dxc += dlo*c0[q] + dhi*c1[q];
        }
      }
      dxa += __shfl_xor(dxa, 16, 32);
      dxb += __shfl_xor(dxb, 16, 32);
      dxc += __shfl_xor(dxc, 16, 32);
      float t0 = 0.f, t1 = 0.f, t2 = 0.f;      // direct term: blended T[i][0:3]
      for (int j = 0; j < 28; ++j) {
        int nn = nb0 + j;
        if (nn < 55) {
          float w = S.logit[m*64 + nn];
          v4f ri = ((const v4f*)(S.tfs + nn * 12))[i];
          t0 += w * ri[0]; t1 += w * ri[1]; t2 += w * ri[2];
        }
      }
      t0 += __shfl_xor(t0, 16, 32);
      t1 += __shfl_xor(t1, 16, 32);
      t2 += __shfl_xor(t2, 16, 32);
      Jm[i*3+0] = t0 + dxa; Jm[i*3+1] = t1 + dxb; Jm[i*3+2] = t2 + dxc;
    }
    __syncthreads();
  }
  float Ji[9]; inv3(Jm, Ji);

  // ---- Broyden (mirrors reference masking exactly, via selects) ----
  float gx0 = e0, gx1 = e1, gx2 = e2;
  float up0 = -(Ji[0]*gx0 + Ji[1]*gx1 + Ji[2]*gx2);
  float up1 = -(Ji[3]*gx0 + Ji[4]*gx1 + Ji[5]*gx2);
  float up2 = -(Ji[6]*gx0 + Ji[7]*gx1 + Ji[8]*gx2);
  float xo0 = x0, xo1 = x1, xo2 = x2;
  float gnopt = sqrtf(gx0*gx0 + gx1*gx1 + gx2*gx2);
  float dx0=0.f,dx1=0.f,dx2=0.f, dg0=0.f,dg1=0.f,dg2=0.f;
  bool idsval = true;

  for (int it = 0; it < 10; ++it) {
    dx0 = idsval ? up0 : dx0; dx1 = idsval ? up1 : dx1; dx2 = idsval ? up2 : dx2;
    x0  = idsval ? x0 + dx0 : x0;
    x1  = idsval ? x1 + dx1 : x1;
    x2  = idsval ? x2 + dx2 : x2;

    float f0, f1, f2, q0, q1, q2;
    g_eval(S, lane, x0, x1, x2, tx, ty, tz, f0, f1, f2, q0, q1, q2);

    dg0 = idsval ? f0 - gx0 : dg0; dg1 = idsval ? f1 - gx1 : dg1; dg2 = idsval ? f2 - gx2 : dg2;
    gx0 = idsval ? f0 : gx0;       gx1 = idsval ? f1 : gx1;       gx2 = idsval ? f2 : gx2;

    float gn = sqrtf(gx0*gx0 + gx1*gx1 + gx2*gx2);
    bool opt = gn < gnopt;
    gnopt = opt ? gn : gnopt;
    xo0 = opt ? x0 : xo0; xo1 = opt ? x1 : xo1; xo2 = opt ? x2 : xo2;
    idsval = (gnopt > 1e-5f) && (gn < 1.0f);

    float vT0 = dx0*Ji[0] + dx1*Ji[3] + dx2*Ji[6];
    float vT1 = dx0*Ji[1] + dx1*Ji[4] + dx2*Ji[7];
    float vT2 = dx0*Ji[2] + dx1*Ji[5] + dx2*Ji[8];
    float bb = vT0*dg0 + vT1*dg1 + vT2*dg2;
    if (fabsf(bb) < 1e-6f) bb = (bb >= 0.f) ? 1e-6f : -1e-6f;
    float Jd0 = Ji[0]*dg0 + Ji[1]*dg1 + Ji[2]*dg2;
    float Jd1 = Ji[3]*dg0 + Ji[4]*dg1 + Ji[5]*dg2;
    float Jd2 = Ji[6]*dg0 + Ji[7]*dg1 + Ji[8]*dg2;
    float rb = 1.f / bb;
    float u0 = (dx0 - Jd0) * rb, u1 = (dx1 - Jd1) * rb, u2 = (dx2 - Jd2) * rb;
    if (idsval) {
      Ji[0] += u0*vT0; Ji[1] += u0*vT1; Ji[2] += u0*vT2;
      Ji[3] += u1*vT0; Ji[4] += u1*vT1; Ji[5] += u1*vT2;
      Ji[6] += u2*vT0; Ji[7] += u2*vT1; Ji[8] += u2*vT2;
    }
    up0 = -(Ji[0]*gx0 + Ji[1]*gx1 + Ji[2]*gx2);
    up1 = -(Ji[3]*gx0 + Ji[4]*gx1 + Ji[5]*gx2);
    up2 = -(Ji[6]*gx0 + Ji[7]*gx1 + Ji[8]*gx2);
  }

  if (lane < 16) {
    out[p*3+0] = xo0; out[p*3+1] = xo1; out[p*3+2] = xo2;
  }
}

// LDS: halves = (3*128 + 64 + WAVES*4*16 + WAVES*16) * WSTR = 768*136 = 104448
//              -> 208896 B
//      floats = 384 + 4*128 + 64 + 660 + 540 + WAVES*1024 + WAVES*64 = 6512
//              -> 26048 B
static constexpr size_t SMEM_BYTES =
    (size_t)(3*128 + 64 + WAVES*4*16 + WAVES*16) * WSTR * 2 + 6512 * 4; // 234944

extern "C" void kernel_launch(void* const* d_in, const int* in_sizes, int n_in,
                              void* d_out, int out_size, void* d_ws, size_t ws_size,
                              hipStream_t stream) {
  (void)in_sizes; (void)n_in; (void)out_size; (void)d_ws; (void)ws_size;
  const float* xd  = (const float*)d_in[0];
  const float* tfs = (const float*)d_in[1];
  const float* W0  = (const float*)d_in[2];  const float* b0 = (const float*)d_in[3];
  const float* W1  = (const float*)d_in[4];  const float* b1 = (const float*)d_in[5];
  const float* W2  = (const float*)d_in[6];  const float* b2 = (const float*)d_in[7];
  const float* W3  = (const float*)d_in[8];  const float* b3 = (const float*)d_in[9];
  const float* Wo  = (const float*)d_in[10]; const float* bo = (const float*)d_in[11];
  float* out = (float*)d_out;

  (void)hipFuncSetAttribute((const void*)deformer_kernel,
                            hipFuncAttributeMaxDynamicSharedMemorySize,
                            (int)SMEM_BYTES);
  deformer_kernel<<<BLOCKS, 128, SMEM_BYTES, stream>>>(
      xd, tfs, W0, b0, W1, b1, W2, b2, W3, b3, Wo, bo, out);
}
